// TRLBlock_16569983828481
// MI455X (gfx1250) — matrix operand order repaired and checked by reading, not compiled, not run on hardware
//
#include <hip/hip_runtime.h>
#include <hip/hip_bf16.h>

#define BB 16
#define TT 8
#define CC 2048
#define HWD 128
#define PP 512
#define RRD 128
#define BNINV 0.9999950000374997f  /* 1/sqrt(1+1e-5) */

typedef __bf16 bf16;
typedef __attribute__((ext_vector_type(16))) __bf16 v16bf;
typedef __attribute__((ext_vector_type(8)))  __bf16 v8bf;
typedef __attribute__((ext_vector_type(8)))  float  v8f;

enum {
  F_RELU  = 1,
  F_SCALE = 2,   // multiply by scale[row]*BNINV (BN gamma)
  F_SHIFT = 4,   // add shift[row] (bias or BN beta)
  F_RES   = 8,   // add resid[idx]
  F_OUTF  = 16,  // write fp32 output
  F_OUTB  = 32,  // write bf16 output
  F_ZMODE = 64   // z[b,row] = mean_col (f1buf - v)^2  (no tensor output)
};

// LDS byte offset of a shared-memory pointer (generic LDS ptr low 32 bits)
__device__ __forceinline__ unsigned lds_off(const void* p) {
  return (unsigned)(unsigned long long)p;
}

// 16-byte async copy global -> LDS (per lane), tracked by ASYNCcnt
__device__ __forceinline__ void async_cp16(unsigned lds, const bf16* g) {
  asm volatile("global_load_async_to_lds_b128 %0, %1, off"
               :: "v"(lds), "v"(g) : "memory");
}

// All activation tensors are stored TRANSPOSED: act[b][n][c], n = h*W+w (HWD=128).
// GEMM: D[b, m, n] = epilogue( sum_k W[m,k] * Xt[b, n, k] )
// grid: (M/128, 1, B), block: 256 threads (8 waves of 32). K-tile = 64.
__global__ __launch_bounds__(256)
void gemm128(const bf16* __restrict__ W, const bf16* __restrict__ X,
             int M, int K,
             const float* __restrict__ scale, const float* __restrict__ shift,
             const float* __restrict__ resid, const float* __restrict__ f1buf,
             float* __restrict__ outf, bf16* __restrict__ outb,
             float* __restrict__ zout, int flags)
{
  __shared__ __align__(16) bf16 As[2][128][64];   // weight tile  [m][k]  (32 KB)
  __shared__ __align__(16) bf16 Xs[2][128][64];   // act tile     [n][k]  (32 KB)

  const int tid     = threadIdx.x;
  const int b       = blockIdx.z;
  const int rowBase = blockIdx.x * 128;
  const bf16* Wt  = W + (size_t)rowBase * K;
  const bf16* Xtb = X + (size_t)b * HWD * K;

  const int lane = tid & 31;
  const int wave = tid >> 5;
  const int lr   = lane & 15;
  const int hi   = lane >> 4;

  v8f zero8 = {0.f,0.f,0.f,0.f,0.f,0.f,0.f,0.f};
  v8f acc[8];
#pragma unroll
  for (int i = 0; i < 8; ++i) acc[i] = zero8;

  // staging: 2 threads/row, 32 contiguous bf16 (four b128 asyncs) each; same for A and X
  const int sr = tid >> 1, sh = (tid & 1) * 32;
  const int arow = wave * 16 + lr;

  auto stage = [&](int buf, int k0) {
    const bf16* aw = Wt  + (size_t)sr * K + k0 + sh;
    const bf16* ax = Xtb + (size_t)sr * K + k0 + sh;
    const unsigned la = lds_off(&As[buf][sr][sh]);
    const unsigned lx = lds_off(&Xs[buf][sr][sh]);
    async_cp16(la,      aw);
    async_cp16(la + 16, aw + 8);
    async_cp16(la + 32, aw + 16);
    async_cp16(la + 48, aw + 24);
    async_cp16(lx,      ax);
    async_cp16(lx + 16, ax + 8);
    async_cp16(lx + 32, ax + 16);
    async_cp16(lx + 48, ax + 24);
  };

  auto compute = [&](int buf) {
#pragma unroll
    for (int ks = 0; ks < 64; ks += 32) {
      // A fragment (16x32 bf16): lanes 0-15 row m, K {0..15}; lanes 16-31 K {16..31}
      v16bf af = *(const v16bf*)&As[buf][arow][ks + hi * 16];
      v16bf bfg[8];
#pragma unroll
      for (int tn = 0; tn < 8; ++tn)
        bfg[tn] = *(const v16bf*)&Xs[buf][tn * 16 + lr][ks + hi * 16];
      __builtin_amdgcn_sched_barrier(0);   // all DS loads issued before WMMAs
#pragma unroll
      for (int tn = 0; tn < 8; ++tn)
        acc[tn] = __builtin_amdgcn_wmma_f32_16x16x32_bf16(
                      false, af, false, bfg[tn], (short)0, acc[tn], false, false);
      __builtin_amdgcn_sched_barrier(0);
    }
  };

  const int nk = K >> 6;
  stage(0, 0);
  for (int kt = 0; kt < nk - 1; ++kt) {           // branch-free steady state
    stage((kt & 1) ^ 1, (kt + 1) << 6);
    asm volatile("s_wait_asynccnt 8" ::: "memory");  // tile kt landed (next 8 in flight)
    __syncthreads();
    compute(kt & 1);
    __syncthreads();
  }
  asm volatile("s_wait_asynccnt 0" ::: "memory");
  __syncthreads();
  compute((nk - 1) & 1);

  // ---- epilogue (outputs transposed: out[b][n][m], m contiguous per lane) ----
  const int rowLoc = rowBase + wave * 16 + hi * 8;  // 8 consecutive rows per lane
  v8f sc8 = zero8, sh8 = zero8;
  if (flags & F_SCALE) {
    sc8 = *(const v8f*)&scale[rowLoc];
#pragma unroll
    for (int r = 0; r < 8; ++r) sc8[r] *= (float)BNINV;
  }
  if (flags & F_SHIFT) sh8 = *(const v8f*)&shift[rowLoc];

  if (flags & F_ZMODE) {
    float s[8];
#pragma unroll
    for (int r = 0; r < 8; ++r) s[r] = 0.0f;
#pragma unroll
    for (int tn = 0; tn < 8; ++tn) {
      const int col = tn * 16 + lr;
      const size_t base = ((size_t)b * HWD + col) * M + rowLoc;
      const v8f f1v = *(const v8f*)&f1buf[base];
      v8f v = acc[tn];
#pragma unroll
      for (int r = 0; r < 8; ++r) {
        float x = v[r];
        if (flags & F_SCALE) x *= sc8[r];
        if (flags & F_SHIFT) x += sh8[r];
        if (flags & F_RELU)  x = fmaxf(x, 0.0f);
        const float d = f1v[r] - x;
        s[r] += d * d;
      }
    }
#pragma unroll
    for (int r = 0; r < 8; ++r) {
      float t = s[r];
      t += __shfl_xor(t, 1, 32);
      t += __shfl_xor(t, 2, 32);
      t += __shfl_xor(t, 4, 32);
      t += __shfl_xor(t, 8, 32);
      if (lr == 0) zout[(size_t)b * M + rowLoc + r] = t * (1.0f / HWD);
    }
  } else {
#pragma unroll
    for (int tn = 0; tn < 8; ++tn) {
      const int col = tn * 16 + lr;
      const size_t base = ((size_t)b * HWD + col) * M + rowLoc;
      v8f v = acc[tn];
      if (flags & F_SCALE) {
#pragma unroll
        for (int r = 0; r < 8; ++r) v[r] *= sc8[r];
      }
      if (flags & F_SHIFT) {
#pragma unroll
        for (int r = 0; r < 8; ++r) v[r] += sh8[r];
      }
      if (flags & F_RES) {
        const v8f rv = *(const v8f*)&resid[base];
#pragma unroll
        for (int r = 0; r < 8; ++r) v[r] += rv[r];
      }
      if (flags & F_RELU) {
#pragma unroll
        for (int r = 0; r < 8; ++r) v[r] = fmaxf(v[r], 0.0f);
      }
      if (flags & F_OUTF) *(v8f*)&outf[base] = v;
      if (flags & F_OUTB) {
        v8bf bv;
#pragma unroll
        for (int r = 0; r < 8; ++r) bv[r] = (bf16)v[r];
        *(v8bf*)&outb[base] = bv;
      }
    }
  }
}

__global__ void k_zero(float* p, int n) {
  int i = blockIdx.x * blockDim.x + threadIdx.x;
  if (i < n) p[i] = 0.0f;
}

__global__ void k_cvt(const float* __restrict__ s, bf16* __restrict__ d, int n) {
  int i = blockIdx.x * blockDim.x + threadIdx.x;
  if (i < n) d[i] = (bf16)s[i];
}

// memo0[b][hw][c] = mean_t xu[b,t,c,hw]  (transposed activation layout)
__global__ void k_memo0(const float* __restrict__ xu, float* __restrict__ memo,
                        bf16* __restrict__ memob) {
  int i = blockIdx.x * blockDim.x + threadIdx.x;
  if (i >= BB * HWD * CC) return;
  int b = i / (HWD * CC), rem = i % (HWD * CC);
  int hw = rem / CC, c = rem % CC;
  const float* src = xu + ((size_t)b * TT * CC + c) * HWD + hw;
  float s = 0.0f;
#pragma unroll
  for (int t = 0; t < TT; ++t) s += src[(size_t)t * CC * HWD];
  s *= (1.0f / TT);
  memo[i] = s;
  memob[i] = (bf16)s;
}

// dst[b][hw][c] = bf16(x[b,t,c,hw])
__global__ void k_slice_bf16(const float* __restrict__ x, bf16* __restrict__ dst, int t) {
  int i = blockIdx.x * blockDim.x + threadIdx.x;
  if (i >= BB * HWD * CC) return;
  int b = i / (HWD * CC), rem = i % (HWD * CC);
  int hw = rem / CC, c = rem % CC;
  dst[i] = (bf16)x[((size_t)(b * TT + t) * CC + c) * HWD + hw];
}

// xsum[b][hw][c] = memo[b][hw][c] + xu[b,t,c,hw]; fp32 + bf16
__global__ void k_xsum(const float* __restrict__ memo, const float* __restrict__ xu, int t,
                       float* __restrict__ xs, bf16* __restrict__ xsb) {
  int i = blockIdx.x * blockDim.x + threadIdx.x;
  if (i >= BB * HWD * CC) return;
  int b = i / (HWD * CC), rem = i % (HWD * CC);
  int hw = rem / CC, c = rem % CC;
  float v = memo[i] + xu[((size_t)(b * TT + t) * CC + c) * HWD + hw];
  xs[i] = v;
  xsb[i] = (bf16)v;
}

// out[i] = mean over 128 contiguous elems of x  (xc spatial means, original layout)
__global__ void k_rowmean(const float* __restrict__ x, float* __restrict__ out, int n) {
  int i = blockIdx.x * blockDim.x + threadIdx.x;
  if (i >= n) return;
  const float* p = x + (size_t)i * HWD;
  float s = 0.0f;
  for (int j = 0; j < HWD; ++j) s += p[j];
  out[i] = s * (1.0f / HWD);
}

// f_unc[b*C+c] += mean_hw memo[b][hw][c]   (strided column mean, coalesced per warp)
__global__ void k_colmean_add(const float* __restrict__ memo, float* __restrict__ out) {
  int i = blockIdx.x * blockDim.x + threadIdx.x;
  if (i >= BB * CC) return;
  int b = i / CC, c = i % CC;
  const float* p = memo + (size_t)b * HWD * CC + c;
  float s = 0.0f;
  for (int hw = 0; hw < HWD; ++hw) s += p[(size_t)hw * CC];
  out[i] += s * (1.0f / HWD);
}

// hidden[b,r] = relu( z[b,:] . Wa1[r,:] )
__global__ void k_se1(const float* __restrict__ z, const float* __restrict__ Wa1,
                      float* __restrict__ hidden) {
  int i = blockIdx.x * blockDim.x + threadIdx.x;
  if (i >= BB * RRD) return;
  int b = i / RRD, r = i % RRD;
  const float* zp = z + (size_t)b * CC;
  const float* wp = Wa1 + (size_t)r * CC;
  float s = 0.0f;
  for (int c = 0; c < CC; ++c) s += zp[c] * wp[c];
  hidden[i] = fmaxf(s, 0.0f);
}

// a = sigmoid(hidden . Wa2[c,:]);  f_corr[b,t,c] += (1+a) * xc_mean[b,t,c]
__global__ void k_se2f(const float* __restrict__ hidden, const float* __restrict__ Wa2,
                       const float* __restrict__ xcm, float* __restrict__ fcorr, int t) {
  int i = blockIdx.x * blockDim.x + threadIdx.x;
  if (i >= BB * CC) return;
  int b = i / CC, c = i % CC;
  const float* hp = hidden + (size_t)b * RRD;
  const float* wp = Wa2 + (size_t)c * RRD;
  float s = 0.0f;
  for (int r = 0; r < RRD; ++r) s += hp[r] * wp[r];
  float a = 1.0f / (1.0f + expf(-s));
  size_t fi = (size_t)(b * TT + t) * CC + c;
  fcorr[fi] += (1.0f + a) * xcm[fi];
}

extern "C" void kernel_launch(void* const* d_in, const int* in_sizes, int n_in,
                              void* d_out, int out_size, void* d_ws, size_t ws_size,
                              hipStream_t stream) {
  const float* xu = (const float*)d_in[0];
  const float* xc = (const float*)d_in[1];

  // ---- carve workspace (256B aligned) ----
  char* wsp = (char*)d_ws;
  auto carve = [&](size_t bytes) -> void* {
    void* p = (void*)wsp;
    wsp += (bytes + 255) & ~(size_t)255;
    return p;
  };
  bf16 *Wf1b[2], *Wf2b[2], *Wc1b[2], *Wc2b[2], *Wc3b[2];
  for (int d = 0; d < 2; ++d) {
    Wf1b[d] = (bf16*)carve((size_t)CC * CC * 2);
    Wf2b[d] = (bf16*)carve((size_t)CC * CC * 2);
    Wc1b[d] = (bf16*)carve((size_t)PP * CC * 2);
    Wc2b[d] = (bf16*)carve((size_t)PP * PP * 2);
    Wc3b[d] = (bf16*)carve((size_t)CC * PP * 2);
  }
  const int NB = BB * HWD * CC;  // 4,194,304
  float* memo   = (float*)carve((size_t)NB * 4);
  bf16*  memob  = (bf16*) carve((size_t)NB * 2);
  float* xsum   = (float*)carve((size_t)NB * 4);
  bf16*  xsumb  = (bf16*) carve((size_t)NB * 2);
  float* f1buf  = (float*)carve((size_t)NB * 4);
  bf16*  xcb    = (bf16*) carve((size_t)NB * 2);
  bf16*  h1     = (bf16*) carve((size_t)BB * HWD * PP * 2);
  bf16*  h2     = (bf16*) carve((size_t)BB * HWD * PP * 2);
  float* z      = (float*)carve((size_t)BB * CC * 4);
  float* hidden = (float*)carve((size_t)BB * RRD * 4);
  float* xcm    = (float*)carve((size_t)BB * TT * CC * 4);

  dim3 blk(256);
#define GRID1(n) dim3((unsigned)(((n) + 255) / 256))

  // ---- convert weights to bf16 (both directions) ----
  for (int d = 0; d < 2; ++d) {
    int ib = 2 + d * 15;
    k_cvt<<<GRID1(CC * CC), blk, 0, stream>>>((const float*)d_in[ib + 0],  Wf1b[d], CC * CC);
    k_cvt<<<GRID1(CC * CC), blk, 0, stream>>>((const float*)d_in[ib + 2],  Wf2b[d], CC * CC);
    k_cvt<<<GRID1(PP * CC), blk, 0, stream>>>((const float*)d_in[ib + 6],  Wc1b[d], PP * CC);
    k_cvt<<<GRID1(PP * PP), blk, 0, stream>>>((const float*)d_in[ib + 9],  Wc2b[d], PP * PP);
    k_cvt<<<GRID1(CC * PP), blk, 0, stream>>>((const float*)d_in[ib + 12], Wc3b[d], CC * PP);
  }

  k_zero<<<GRID1(out_size), blk, 0, stream>>>((float*)d_out, out_size);
  k_rowmean<<<GRID1(BB * TT * CC), blk, 0, stream>>>(xc, xcm, BB * TT * CC);

  float* f_unc  = (float*)d_out;             // [B,C]
  float* f_corr = (float*)d_out + BB * CC;   // [B,T,C]

  const dim3 g16(16, 1, BB), g4(4, 1, BB);

  for (int d = 0; d < 2; ++d) {
    int ib = 2 + d * 15;
    const float* bf1 = (const float*)d_in[ib + 1];
    const float* bf2 = (const float*)d_in[ib + 3];
    const float* Wa1 = (const float*)d_in[ib + 4];
    const float* Wa2 = (const float*)d_in[ib + 5];
    const float* g1  = (const float*)d_in[ib + 7];
    const float* be1 = (const float*)d_in[ib + 8];
    const float* g2  = (const float*)d_in[ib + 10];
    const float* be2 = (const float*)d_in[ib + 11];
    const float* g3  = (const float*)d_in[ib + 13];
    const float* be3 = (const float*)d_in[ib + 14];

    k_memo0<<<GRID1(NB), blk, 0, stream>>>(xu, memo, memob);

    for (int s = 0; s < TT; ++s) {
      int t = d ? (TT - 1 - s) : s;
      k_slice_bf16<<<GRID1(NB), blk, 0, stream>>>(xc, xcb, t);

      // f1 = relu(Wf1 x memo + bf1)   (stored transposed)
      gemm128<<<g16, blk, 0, stream>>>(Wf1b[d], memob, CC, CC,
          nullptr, bf1, nullptr, nullptr, f1buf, nullptr, nullptr,
          F_SHIFT | F_RELU | F_OUTF);
      // z = mean_hw (f1 - relu(Wf2 x xc + bf2))^2
      gemm128<<<g16, blk, 0, stream>>>(Wf2b[d], xcb, CC, CC,
          nullptr, bf2, nullptr, f1buf, nullptr, nullptr, z,
          F_SHIFT | F_RELU | F_ZMODE);

      k_se1<<<GRID1(BB * RRD), blk, 0, stream>>>(z, Wa1, hidden);
      k_se2f<<<GRID1(BB * CC), blk, 0, stream>>>(hidden, Wa2, xcm, f_corr, t);

      // basic block
      k_xsum<<<GRID1(NB), blk, 0, stream>>>(memo, xu, t, xsum, xsumb);
      gemm128<<<g4, blk, 0, stream>>>(Wc1b[d], xsumb, PP, CC,
          g1, be1, nullptr, nullptr, nullptr, h1, nullptr,
          F_SCALE | F_SHIFT | F_RELU | F_OUTB);
      gemm128<<<g4, blk, 0, stream>>>(Wc2b[d], h1, PP, PP,
          g2, be2, nullptr, nullptr, nullptr, h2, nullptr,
          F_SCALE | F_SHIFT | F_RELU | F_OUTB);
      gemm128<<<g16, blk, 0, stream>>>(Wc3b[d], h2, CC, PP,
          g3, be3, xsum, nullptr, memo, memob, nullptr,
          F_SCALE | F_SHIFT | F_RES | F_RELU | F_OUTF | F_OUTB);
    }
    k_colmean_add<<<GRID1(BB * CC), blk, 0, stream>>>(memo, f_unc);
  }
#undef GRID1
}
